// MultiHeadAttention_81415400063613
// MI455X (gfx1250) — matrix operand order repaired and checked by reading, not compiled, not run on hardware
//
#include <hip/hip_runtime.h>
#include <cstdint>

// ---------------- types ----------------
typedef __attribute__((ext_vector_type(16))) __bf16 bf16x16;
typedef __attribute__((ext_vector_type(8)))  float  f32x8;

union Frag16 { unsigned int u[8]; bf16x16 v; };

__device__ __forceinline__ unsigned short f2bf(float f) {
  unsigned int u = __float_as_uint(f);
  u += 0x7FFFu + ((u >> 16) & 1u);          // round-to-nearest-even
  return (unsigned short)(u >> 16);
}
__device__ __forceinline__ float bf2f(unsigned short h) {
  return __uint_as_float(((unsigned int)h) << 16);
}

// ---------------- problem constants ----------------
constexpr int NHEADS = 8;
constexpr int SEQL   = 10;
constexpr int BSZ    = 1024;     // = 32*32
constexpr int KDIM   = 256;      // = V_DIM
constexpr int EMBD   = 2048;
constexpr int MROWS  = BSZ * SEQL;   // 10240
constexpr float NORM = 0.0625f;      // 1/sqrt(256)

// ---------------- conversions ----------------
__global__ void conv_f32_bf16(const float* __restrict__ in,
                              unsigned short* __restrict__ out, long n) {
  long i = (long)blockIdx.x * blockDim.x + threadIdx.x;
  if (i < n) out[i] = f2bf(in[i]);
}

// in: (8, K, P) f32  ->  out: (K, 8*P) bf16  (B operand, K x N row-major)
__global__ void conv_weight(const float* __restrict__ in,
                            unsigned short* __restrict__ out, int K, int P) {
  long n = (long)NHEADS * K * P;
  long i = (long)blockIdx.x * blockDim.x + threadIdx.x;
  if (i >= n) return;
  int j = (int)(i % P);
  long r = i / P;
  int k = (int)(r % K);
  int h = (int)(r / K);
  out[(long)k * (NHEADS * P) + (long)h * P + j] = f2bf(in[i]);
}

// ---------------- bf16 WMMA GEMM ----------------
// C(MxN) = A(MxK, bf16 rm) * B(KxN, bf16 rm), f32 accumulate.
// Block tile 128x128, K-step 32.  8 waves: 2(M-halves) x 4(N-quarters);
// wave tile 64x32 = 4x2 WMMA 16x16 tiles.
// A panel staged with GLOBAL_LOAD_ASYNC_TO_LDS_B128 (ASYNCcnt path);
// B panel staged via VGPRs because it is repacked into K-pair dwords.
#define BM 128
#define BN 128
#define BK 32
#define ASTR 20     // dwords per A-row in LDS (80B, 16B aligned, conflict-free)
#define BSTR 132    // dwords per packed-B row (528B, 16B aligned)
#define A_LDS_DW (BM * ASTR)          // 2560 dwords
#define B_LDS_DW (16 * BSTR)          // 2112 dwords

__global__ __launch_bounds__(256) void wmma_gemm_bf16(
    const unsigned short* __restrict__ A,
    const unsigned short* __restrict__ B,
    void* __restrict__ outp,
    int M, int N, int K, int mode)   // mode 0: q/k f32; 1: v bf16; 2: plain f32
{
  // single LDS block: As at LDS byte offset 0, Bp right after (async VDST
  // operand needs the raw LDS byte offset).
  __shared__ unsigned int lds_mem[A_LDS_DW + B_LDS_DW];
  unsigned int* As = lds_mem;
  unsigned int* Bp = lds_mem + A_LDS_DW;

  const int t    = threadIdx.x;
  const int lane = t & 31;
  const int wave = t >> 5;
  const int wm   = wave >> 2;     // 0..1
  const int wn   = wave & 3;      // 0..3
  const long bm0 = (long)blockIdx.y * BM;
  const long bn0 = (long)blockIdx.x * BN;

  f32x8 acc[4][2];
  const f32x8 zero = {0.f,0.f,0.f,0.f,0.f,0.f,0.f,0.f};
  #pragma unroll
  for (int i = 0; i < 4; ++i)
    #pragma unroll
    for (int j = 0; j < 2; ++j) acc[i][j] = zero;

  for (int k0 = 0; k0 < K; k0 += BK) {
    // --- stage A panel (128 x 32 halves) via async global->LDS DMA ---
    #pragma unroll
    for (int cc = 0; cc < 2; ++cc) {
      int c   = t + cc * 256;
      int row = c >> 2;
      int seg = c & 3;
      const unsigned short* gp = A + (bm0 + row) * (long)K + k0 + seg * 8;
      unsigned int lds_off = (unsigned int)(row * (ASTR * 4) + seg * 16); // bytes
      asm volatile("global_load_async_to_lds_b128 %0, %1, off"
                   :
                   : "v"(lds_off), "v"((unsigned long)gp)
                   : "memory");
      if (k0 + BK < K) __builtin_prefetch((const void*)(gp + BK), 0, 1);
    }
    // --- stage B panel packed as K-pairs: Bp[kp][col] = {B[2kp][c], B[2kp+1][c]} ---
    {
      int tk   = (t >> 4) * 2;
      int tcol = (t & 15) * 8;
      uint4 lo4 = *(const uint4*)(B + (long)(k0 + tk)     * N + bn0 + tcol);
      uint4 hi4 = *(const uint4*)(B + (long)(k0 + tk + 1) * N + bn0 + tcol);
      const unsigned int* lo = (const unsigned int*)&lo4;
      const unsigned int* hi = (const unsigned int*)&hi4;
      unsigned int pk[8];
      #pragma unroll
      for (int j = 0; j < 4; ++j) {
        pk[2*j]   = (lo[j] & 0xffffu) | ((hi[j] & 0xffffu) << 16);
        pk[2*j+1] = (lo[j] >> 16)     |  (hi[j] & 0xffff0000u);
      }
      int base = (t >> 4) * BSTR + tcol;
      uint4 s0; s0.x = pk[0]; s0.y = pk[1]; s0.z = pk[2]; s0.w = pk[3];
      uint4 s1; s1.x = pk[4]; s1.y = pk[5]; s1.z = pk[6]; s1.w = pk[7];
      *(uint4*)&Bp[base]     = s0;
      *(uint4*)&Bp[base + 4] = s1;
    }
    // wait for this wave's async copies, then block barrier
    asm volatile("s_wait_asynccnt 0x0" ::: "memory");
    __syncthreads();

    // --- fragments in contiguous K order, k = 16*(lane/16) + s, same for A and B ---
    Frag16 af[4], bfr[2];
    {
      int arow = wm * 64 + (lane & 15);
      int kdb  = (lane < 16) ? 0 : 8;   // lanes 0-15: K 0..15; lanes 16-31: K 16..31
      #pragma unroll
      for (int mt = 0; mt < 4; ++mt) {
        int rb = (arow + mt * 16) * ASTR;
        #pragma unroll
        for (int r = 0; r < 4; ++r) {
          af[mt].u[r]     = As[rb + kdb + r];
          af[mt].u[4 + r] = As[rb + kdb + 4 + r];
        }
      }
      int bcol = wn * 32 + (lane & 15);
      int kpb  = (lane < 16) ? 0 : 8;   // lanes 0-15: K 0..15; lanes 16-31: K 16..31
      #pragma unroll
      for (int nt = 0; nt < 2; ++nt)
        #pragma unroll
        for (int r = 0; r < 8; ++r)
          bfr[nt].u[r] = Bp[(kpb + r) * BSTR + bcol + nt * 16];
    }
    #pragma unroll
    for (int mt = 0; mt < 4; ++mt)
      #pragma unroll
      for (int nt = 0; nt < 2; ++nt)
        acc[mt][nt] = __builtin_amdgcn_wmma_f32_16x16x32_bf16(
            false, af[mt].v, false, bfr[nt].v, (short)0, acc[mt][nt], false, false);
    __syncthreads();
  }

  // --- epilogue: C tile layout: lanes 0-15 -> M=r, lanes 16-31 -> M=8+r; N=lane&15 ---
  const int nloc = lane & 15;
  const int mhi  = (lane >> 4) * 8;
  #pragma unroll
  for (int mt = 0; mt < 4; ++mt) {
    #pragma unroll
    for (int nt = 0; nt < 2; ++nt) {
      long gn = bn0 + wn * 32 + nt * 16 + nloc;
      #pragma unroll
      for (int r = 0; r < 8; ++r) {
        long gm = bm0 + wm * 64 + mt * 16 + mhi + r;
        float cv = acc[mt][nt][r];
        if (mode == 2) {
          ((float*)outp)[gm * N + gn] = cv;
        } else if (mode == 0) {
          int h = (int)(gn >> 8), d = (int)(gn & 255);
          int b = (int)(gm / 10), s = (int)(gm % 10);
          ((float*)outp)[(((long)h * BSZ + b) * SEQL + s) * KDIM + d] = cv;
        } else {
          int h = (int)(gn / 768), j = (int)(gn % 768);
          int c = j >> 8, d = j & 255;
          int b = (int)(gm / 10), s = (int)(gm % 10);
          int kv = s * 3 + c;   // v reshape is s-major over (SEQ*K_NUM)
          ((unsigned short*)outp)[(((long)h * BSZ + b) * 30 + kv) * KDIM + d] = f2bf(cv);
        }
      }
    }
  }
}

// ---------------- deltas = q2 @ dw + db  (q2 = scrambled q) ----------------
__global__ __launch_bounds__(64) void deltas_kernel(
    const float* __restrict__ q,    // (8,1024,10,256)
    const float* __restrict__ dw,   // (2560,60)
    const float* __restrict__ db,   // (60)
    float* __restrict__ deltas)     // (8,1024,60)
{
  __shared__ float row[2560];
  const int t = threadIdx.x;
  const long rb = (long)blockIdx.x * 2560;
  for (int j = t; j < 2560; j += 64) {
    long f = rb + j;                 // flat idx into transpose(q,(1,2,3,0))
    int hh = (int)(f & 7);
    int dd = (int)((f >> 3) & 255);
    long r2 = f >> 11;
    int ss = (int)(r2 % 10);
    int bb = (int)(r2 / 10);
    row[j] = q[(((long)hh * BSZ + bb) * SEQL + ss) * KDIM + dd];
  }
  __syncthreads();
  if (t < 60) {
    float acc = db[t];
    for (int j = 0; j < 2560; ++j) acc += row[j] * dw[(long)j * 60 + t];
    deltas[(long)blockIdx.x * 60 + t] = acc;
  }
}

// ---------------- gx,gy: phi = ref + d, L2-normalized over H (wave32 reduce) ----------------
__global__ __launch_bounds__(32) void grid_kernel(
    const float* __restrict__ deltas,  // (8,1024,60)
    float* __restrict__ gg)            // (30,8,1024,2)
{
  int blk = blockIdx.x;                // ((kn*10+s)*8 + h2)*32 + x
  int x  = blk & 31;
  int h2 = (blk >> 5) & 7;
  int s  = (blk >> 8) % 10;
  int kn = blk / 2560;
  int y  = threadIdx.x;                // 32 lanes = 32 H values
  long di = ((long)h2 * BSZ + y * 32 + x) * 60 + s * 6 + kn * 2;
  float px = (float)x * (1.0f / 31.0f) + deltas[di + 0];
  float py = (float)y * (1.0f / 31.0f) + deltas[di + 1];
  float sx = px * px, sy = py * py;
  #pragma unroll
  for (int m = 16; m > 0; m >>= 1) {
    sx += __shfl_xor(sx, m, 32);
    sy += __shfl_xor(sy, m, 32);
  }
  float nx = fmaxf(sqrtf(sx), 1e-12f);
  float ny = fmaxf(sqrtf(sy), 1e-12f);
  long gi = (((long)(kn * 10 + s) * NHEADS + h2) * BSZ + y * 32 + x) * 2;
  gg[gi + 0] = px / nx;
  gg[gi + 1] = py / ny;
}

// ---------------- fused attention: bilinear gather -> qk -> softmax -> attn@v ----------------
__global__ __launch_bounds__(256) void attn_kernel(
    const float* __restrict__ q,           // (8,1024,10,256)
    const float* __restrict__ kf,          // (8,1024,10,256)
    const unsigned short* __restrict__ v,  // (8,1024,30,256) bf16
    const float* __restrict__ gg,          // (30,8,1024,2)
    float* __restrict__ attnpart,          // (8,1024,30)
    unsigned short* __restrict__ headflat) // (10240,2048) bf16
{
  __shared__ float qs[SEQL * KDIM];
  __shared__ float fs[30 * KDIM];
  __shared__ float tw[30][4];
  __shared__ int   tb[30][4];
  __shared__ float at[SEQL][32];

  const int t  = threadIdx.x;
  const int hb = blockIdx.x;       // h*1024 + b
  const int h  = hb >> 10;
  const int b  = hb & 1023;

  // q3 row (scrambled q2) into LDS
  for (int i = t; i < SEQL * KDIM; i += 256) {
    long f = (long)hb * 2560 + i;
    int hh = (int)(f & 7);
    int dd = (int)((f >> 3) & 255);
    long r2 = f >> 11;
    int ss = (int)(r2 % 10);
    int bb = (int)(r2 / 10);
    qs[i] = q[(((long)hh * BSZ + bb) * SEQL + ss) * KDIM + dd];
  }
  // bilinear taps for the 30 (kn,s) sample points of this (h,b)
  if (t < 30) {
    long gi = ((long)t * NHEADS + h) * BSZ + b;
    float gx = gg[gi * 2 + 0];
    float gy = gg[gi * 2 + 1];
    float xf = ((gx + 1.0f) * 32.0f - 1.0f) * 0.5f;
    float yf = ((gy + 1.0f) * 32.0f - 1.0f) * 0.5f;
    float x0 = floorf(xf), y0 = floorf(yf);
    float x1 = x0 + 1.0f,  y1 = y0 + 1.0f;
    float wx1 = xf - x0, wx0 = 1.0f - wx1;
    float wy1 = yf - y0, wy0 = 1.0f - wy1;
    float xs[4] = {x0, x1, x0, x1};
    float ys[4] = {y0, y0, y1, y1};
    float ws[4] = {wy0*wx0, wy0*wx1, wy1*wx0, wy1*wx1};
    #pragma unroll
    for (int i2 = 0; i2 < 4; ++i2) {
      bool valid = (xs[i2] >= 0.f) && (xs[i2] < 32.f) && (ys[i2] >= 0.f) && (ys[i2] < 32.f);
      int xc = min(max((int)xs[i2], 0), 31);
      int yc = min(max((int)ys[i2], 0), 31);
      tb[t][i2] = yc * 32 + xc;
      tw[t][i2] = valid ? ws[i2] : 0.0f;
    }
  }
  __syncthreads();

  // feat rows: weighted gather of k-feature rows
  for (int i = t; i < 30 * KDIM; i += 256) {
    int kk = i >> 8, d = i & 255;
    int sq = kk % 10;
    float acc = 0.f;
    #pragma unroll
    for (int i2 = 0; i2 < 4; ++i2)
      acc += tw[kk][i2] * kf[(((long)h * BSZ + tb[kk][i2]) * SEQL + sq) * KDIM + d];
    fs[i] = acc;
  }
  __syncthreads();

  // qk (10 x 30)
  for (int p = t; p < 300; p += 256) {
    int s = p / 30, kk = p % 30;
    float acc = 0.f;
    for (int d = 0; d < KDIM; ++d) acc += qs[s * KDIM + d] * fs[kk * KDIM + d];
    at[s][kk] = acc * NORM;
  }
  __syncthreads();

  // softmax per row
  if (t < SEQL) {
    float m = -1e30f;
    for (int kk = 0; kk < 30; ++kk) m = fmaxf(m, at[t][kk]);
    float e[30], sum = 0.f;
    for (int kk = 0; kk < 30; ++kk) { e[kk] = expf(at[t][kk] - m); sum += e[kk]; }
    float inv = 1.0f / sum;
    for (int kk = 0; kk < 30; ++kk) at[t][kk] = e[kk] * inv;
  }
  __syncthreads();

  if (t < 30) attnpart[(long)hb * 30 + t] = at[0][t];

  // head = attn @ v ; write head_flat (b*10+s, h*256+d) as bf16
  for (int i = t; i < SEQL * KDIM; i += 256) {
    int s = i >> 8, d = i & 255;
    float acc = 0.f;
    #pragma unroll
    for (int kk = 0; kk < 30; ++kk)
      acc += at[s][kk] * bf2f(v[((long)hb * 30 + kk) * KDIM + d]);
    headflat[((long)b * SEQL + s) * EMBD + h * KDIM + d] = f2bf(acc);
  }
}

// ---------------- attn_map = mean over heads of attn[:, :, 0, :] ----------------
__global__ void attnmap_reduce(const float* __restrict__ part,
                               float* __restrict__ outm) {
  int idx = blockIdx.x * blockDim.x + threadIdx.x;
  if (idx >= BSZ * 30) return;
  int b = idx / 30, kk = idx % 30;
  float s = 0.f;
  #pragma unroll
  for (int h = 0; h < NHEADS; ++h) s += part[((long)h * BSZ + b) * 30 + kk];
  outm[idx] = s * 0.125f;
}

// ---------------- launcher ----------------
extern "C" void kernel_launch(void* const* d_in, const int* in_sizes, int n_in,
                              void* d_out, int out_size, void* d_ws, size_t ws_size,
                              hipStream_t stream) {
  (void)in_sizes; (void)n_in; (void)out_size; (void)ws_size;
  const float* x  = (const float*)d_in[0];
  const float* Wq = (const float*)d_in[1];
  const float* Wk = (const float*)d_in[2];
  const float* Wv = (const float*)d_in[3];
  const float* Wo = (const float*)d_in[4];
  const float* dw = (const float*)d_in[5];
  const float* db = (const float*)d_in[6];

  char* ws = (char*)d_ws;
  unsigned short* xbf = (unsigned short*)(ws + 0L);          // 10240x2048 bf16
  unsigned short* wqb = (unsigned short*)(ws + 41943040L);   // 2048x2048 bf16
  unsigned short* wkb = (unsigned short*)(ws + 50331648L);   // 2048x2048 bf16
  unsigned short* wvb = (unsigned short*)(ws + 58720256L);   // 2048x6144 bf16
  unsigned short* wob = (unsigned short*)(ws + 83886080L);   // 2048x2048 bf16
  float*          qf  = (float*)         (ws + 92274688L);   // (8,1024,10,256) f32
  float*          kfp = (float*)         (ws + 176160768L);  // (8,1024,10,256) f32
  unsigned short* vbf = (unsigned short*)(ws + 260046848L);  // (8,1024,30,256) bf16
  float*          del = (float*)         (ws + 385875968L);  // (8,1024,60)
  float*          gg  = (float*)         (ws + 387842048L);  // (30,8,1024,2)
  float*          ap  = (float*)         (ws + 389808128L);  // (8,1024,30)
  unsigned short* hfb = (unsigned short*)(ws + 390791168L);  // (10240,2048) bf16

  float* out_main = (float*)d_out;                 // (1024,10,2048)
  float* out_attn = out_main + 20971520L;          // (1024,30)

  // conversions
  {
    long n = (long)MROWS * EMBD;
    conv_f32_bf16<<<(unsigned)((n + 255) / 256), 256, 0, stream>>>(x, xbf, n);
  }
  {
    long n = (long)EMBD * EMBD;
    conv_f32_bf16<<<(unsigned)((n + 255) / 256), 256, 0, stream>>>(Wo, wob, n);
  }
  {
    long n = (long)NHEADS * 2048 * 256;
    conv_weight<<<(unsigned)((n + 255) / 256), 256, 0, stream>>>(Wq, wqb, 2048, 256);
    conv_weight<<<(unsigned)((n + 255) / 256), 256, 0, stream>>>(Wk, wkb, 2048, 256);
  }
  {
    long n = (long)NHEADS * 2048 * 768;
    conv_weight<<<(unsigned)((n + 255) / 256), 256, 0, stream>>>(Wv, wvb, 2048, 768);
  }

  // projections (WMMA)
  wmma_gemm_bf16<<<dim3(2048 / BN, MROWS / BM), 256, 0, stream>>>(xbf, wqb, qf,  MROWS, 2048, 2048, 0);
  wmma_gemm_bf16<<<dim3(2048 / BN, MROWS / BM), 256, 0, stream>>>(xbf, wkb, kfp, MROWS, 2048, 2048, 0);
  wmma_gemm_bf16<<<dim3(6144 / BN, MROWS / BM), 256, 0, stream>>>(xbf, wvb, vbf, MROWS, 6144, 2048, 1);

  // deformable sampling pipeline
  deltas_kernel<<<NHEADS * BSZ, 64, 0, stream>>>(qf, dw, db, del);
  grid_kernel<<<3 * 10 * 8 * 32, 32, 0, stream>>>(del, gg);
  attn_kernel<<<NHEADS * BSZ, 256, 0, stream>>>(qf, kfp, vbf, gg, ap, hfb);
  attnmap_reduce<<<(BSZ * 30 + 255) / 256, 256, 0, stream>>>(ap, out_attn);

  // output projection (WMMA) -> d_out row-major
  wmma_gemm_bf16<<<dim3(2048 / BN, MROWS / BM), 256, 0, stream>>>(hfb, wob, out_main, MROWS, 2048, 2048, 2);
}
